// RNN_52578989638373
// MI455X (gfx1250) — matrix operand order repaired and checked by reading, not compiled, not run on hardware
//
#include <hip/hip_runtime.h>
#include <hip/hip_bf16.h>
#include <math.h>

// ---------------------------------------------------------------------------
// RNN: out = head( scan_t( tanh(xw_t + h @ W_h^T) ) )
// B=64, T=1024, EMB=512, VOCAB=256.  bf16 WMMA (v_wmma_f32_16x16x32_bf16),
// fp32 accumulate.  Weights W[o,e] row-major => B-fragments are contiguous
// rows of W (no transpose loads needed).  Scan kernel prefetches the xw_t
// tile into LDS with GLOBAL_LOAD_ASYNC_TO_LDS (ASYNCcnt) when available.
// ---------------------------------------------------------------------------

typedef __bf16 bf16_t;
typedef __attribute__((ext_vector_type(16))) __bf16 v16bf;
typedef __attribute__((ext_vector_type(8)))  __bf16 v8bf;   // 16 bytes
typedef __attribute__((ext_vector_type(8)))  float  v8f;
typedef __attribute__((ext_vector_type(4)))  int    v4i;    // 16 bytes

constexpr int kEMB   = 512;
constexpr int kVOCAB = 256;
constexpr int kB     = 64;
constexpr int kT     = 1024;
constexpr int kRows  = kB * kT;      // 65536 GEMM rows
constexpr int kNWG   = 16;           // scan workgroups; each owns 32 out cols

#if defined(__AMDGCN__) && __has_builtin(__builtin_amdgcn_global_load_async_to_lds_b128) && \
    __has_builtin(__builtin_amdgcn_s_wait_asynccnt)
#define USE_ASYNC_LDS 1
typedef __attribute__((address_space(1))) v4i as1_v4i;   // global int4*
typedef __attribute__((address_space(3))) v4i as3_v4i;   // LDS int4*
#else
#define USE_ASYNC_LDS 0
#endif

union FragBF { v16bf v; struct { v8bf lo; v8bf hi; } p; };

// A fragment 16x32 bf16 from row-major source. `row` = per-lane row pointer
// (row index = mbase + (lane&15)).  ISA layout: lane-half hi holds
// K = k0 + hi*8 + [0..7]  (VGPR0..3) and K = k0 + 16 + hi*8 + [0..7] (VGPR4..7).
__device__ __forceinline__ v16bf load_frag_a(const bf16_t* row, int k0, int hi) {
    FragBF f;
    f.p.lo = *(const v8bf*)(row + k0 + hi * 8);
    f.p.hi = *(const v8bf*)(row + k0 + 16 + hi * 8);
    return f.v;
}

// B fragment 32x16 where B[k][n] = W[n][k], W row-major stride kEMB.
// `wrow` = W + n*kEMB with n = nbase + (lane&15).  Per lane: K = k0 + hi*8 + [0..7]
// and K = k0 + 16 + hi*8 + [0..7] (same order as A).
__device__ __forceinline__ v16bf load_frag_b(const bf16_t* wrow, int k0, int hi) {
    FragBF f;
    f.p.lo = *(const v8bf*)(wrow + k0 + hi * 8);
    f.p.hi = *(const v8bf*)(wrow + k0 + 16 + hi * 8);
    return f.v;
}

__device__ __forceinline__ v8f wmma_bf16(v16bf a, v16bf b, v8f c) {
    // (neg_a, A, neg_b, B, c_mod, C, reuse_a, reuse_b)
    return __builtin_amdgcn_wmma_f32_16x16x32_bf16(false, a, false, b, (short)0, c,
                                                   false, false);
}

// 16-byte chunk copy global -> LDS; async path uses ASYNCcnt.
__device__ __forceinline__ void copy16_g2l(const float* g, float* l) {
#if USE_ASYNC_LDS
    __builtin_amdgcn_global_load_async_to_lds_b128((as1_v4i*)(g), (as3_v4i*)(l), 0, 0);
#else
    *(float4*)l = *(const float4*)g;
#endif
}

__device__ __forceinline__ void wait_async_lds() {
#if USE_ASYNC_LDS
    __builtin_amdgcn_s_wait_asynccnt(0);
#endif
}

// ---------------------------------------------------------------------------
// Kernel 0: convert weights/embedding to bf16, init h0 buffer, reset barrier.
// ---------------------------------------------------------------------------
__global__ void prep_kernel(const float* __restrict__ emb,
                            const float* __restrict__ w_cell,
                            const float* __restrict__ w_head,
                            const float* __restrict__ h0,
                            bf16_t* __restrict__ embb,
                            bf16_t* __restrict__ wx,
                            bf16_t* __restrict__ wh,
                            bf16_t* __restrict__ wheadb,
                            bf16_t* __restrict__ Hbuf,
                            unsigned* __restrict__ barrier_cnt) {
    const int idx0   = blockIdx.x * blockDim.x + threadIdx.x;
    const int stride = gridDim.x * blockDim.x;
    if (idx0 == 0) *barrier_cnt = 0u;
    for (int i = idx0; i < kVOCAB * kEMB; i += stride) embb[i] = (bf16_t)emb[i];
    for (int i = idx0; i < kEMB * kEMB; i += stride) {
        const int o = i >> 9, e = i & (kEMB - 1);
        wx[i] = (bf16_t)w_cell[o * (2 * kEMB) + e];
        wh[i] = (bf16_t)w_cell[o * (2 * kEMB) + kEMB + e];
    }
    for (int i = idx0; i < kVOCAB * kEMB; i += stride) wheadb[i] = (bf16_t)w_head[i];
    for (int i = idx0; i < kB * kEMB; i += stride) Hbuf[i] = (bf16_t)h0[i & (kEMB - 1)];
}

// ---------------------------------------------------------------------------
// Kernel 1: xw[t*64+b][o] = sum_e emb[x[b][t]][e] * w_x[o][e] + b_cell[o]
// Workgroup (8 waves) = 128 rows x 64 cols; wave = 16 rows x 64 cols.
// Double-buffered K loop so fragment loads overlap WMMA issue.
// ---------------------------------------------------------------------------
__global__ __launch_bounds__(256) void xw_gemm_kernel(
    const int* __restrict__ x, const bf16_t* __restrict__ embb,
    const bf16_t* __restrict__ wx, const float* __restrict__ bcell,
    float* __restrict__ xw) {
    const int wave = threadIdx.x >> 5;
    const int lane = threadIdx.x & 31;
    const int l15 = lane & 15, hi = lane >> 4;
    const int mtile = blockIdx.y * 128 + wave * 16;
    const int ntile = blockIdx.x * 64;

    // A row for this lane: GEMM row m = t*64 + b (xw is [t][b][o])
    const int m = mtile + l15;
    const int b = m & (kB - 1);
    const int t = m >> 6;
    const int tok = x[b * kT + t];
    const bf16_t* arow  = embb + (size_t)tok * kEMB;
    const bf16_t* brow0 = wx + (size_t)(ntile + l15) * kEMB;

    v8f acc[4] = {};
    v16bf a0 = load_frag_a(arow, 0, hi);
    v16bf b0[4];
#pragma unroll
    for (int j = 0; j < 4; ++j) b0[j] = load_frag_b(brow0 + (size_t)j * 16 * kEMB, 0, hi);

#pragma unroll
    for (int k0 = 32; k0 < kEMB; k0 += 32) {
        const v16bf a1 = load_frag_a(arow, k0, hi);
        v16bf b1[4];
#pragma unroll
        for (int j = 0; j < 4; ++j)
            b1[j] = load_frag_b(brow0 + (size_t)j * 16 * kEMB, k0, hi);
#pragma unroll
        for (int j = 0; j < 4; ++j) acc[j] = wmma_bf16(a0, b0[j], acc[j]);
        a0 = a1;
#pragma unroll
        for (int j = 0; j < 4; ++j) b0[j] = b1[j];
    }
#pragma unroll
    for (int j = 0; j < 4; ++j) acc[j] = wmma_bf16(a0, b0[j], acc[j]);

#pragma unroll
    for (int j = 0; j < 4; ++j) {
        const int n = ntile + j * 16 + l15;
        const float bias = bcell[n];
#pragma unroll
        for (int v = 0; v < 8; ++v) {
            const int mm = mtile + hi * 8 + v;
            xw[(size_t)mm * kEMB + n] = acc[j][v] + bias;
        }
    }
}

// ---------------------------------------------------------------------------
// Kernel 2: persistent recurrent scan.  16 WGs, each owns 32 output columns;
// W_h slice (32x512 bf16 = 32KB) in LDS; xw_t tile (64x32 fp32 = 8KB)
// double-buffered in LDS via async global->LDS prefetch.  h state (64x512
// bf16) ping-pongs through L2; device-scope atomic barrier per step.
// ---------------------------------------------------------------------------
__global__ __launch_bounds__(256) void rnn_scan_kernel(
    const bf16_t* __restrict__ wh, const float* __restrict__ xw,
    bf16_t* __restrict__ Hbuf, bf16_t* __restrict__ hs,
    unsigned* __restrict__ barrier_cnt) {
    __shared__ bf16_t lds_wh[32 * kEMB];      // 32 KB
    __shared__ float  lds_xw[2][kB * 32];     // 2 x 8 KB ping-pong

    const int wg = blockIdx.x;   // 0..15
    const int o0 = wg * 32;      // this WG's first output column

    // ---- preload W_h slice (async if available) ----
    {
        const float* src = (const float*)(wh + (size_t)o0 * kEMB);  // 16B chunks
        float* dst = (float*)lds_wh;
        // 32KB = 2048 x 16B chunks, 256 threads -> 8 each
#pragma unroll
        for (int i = 0; i < 8; ++i) {
            const int c = threadIdx.x + i * 256;
            copy16_g2l(src + c * 4, dst + c * 4);
        }
    }
    // ---- prefetch xw tile for t=0 ----
    {
#pragma unroll
        for (int i = 0; i < 2; ++i) {
            const int c = threadIdx.x + i * 256;       // 512 chunks of 16B = 8KB
            const int row = c >> 3;                    // 8 chunks per 128B row
            const int ce  = (c & 7) * 4;               // float offset within row
            copy16_g2l(xw + (size_t)row * kEMB + o0 + ce, &lds_xw[0][c * 4]);
        }
    }
    wait_async_lds();
    __syncthreads();

    const int wave = threadIdx.x >> 5;   // 0..7
    const int lane = threadIdx.x & 31;
    const int l15 = lane & 15, hi = lane >> 4;
    const int mbase = (wave >> 1) * 16;          // batch-row subtile (4 of them)
    const int nloc  = (wave & 1) * 16 + l15;     // local column 0..31
    const int n     = o0 + nloc;                 // global output column
    const bf16_t* bptr = lds_wh + (size_t)nloc * kEMB;

    bf16_t* Hcur = Hbuf;
    bf16_t* Hnxt = Hbuf + kB * kEMB;
    int cur = 0;

    for (int t = 0; t < kT; ++t) {
        // ---- async prefetch next step's xw tile into the other LDS buffer ----
        if (t + 1 < kT) {
            const float* xwn = xw + (size_t)(t + 1) * kB * kEMB;
#pragma unroll
            for (int i = 0; i < 2; ++i) {
                const int c = threadIdx.x + i * 256;
                const int row = c >> 3;
                const int ce  = (c & 7) * 4;
                copy16_g2l(xwn + (size_t)row * kEMB + o0 + ce,
                           &lds_xw[cur ^ 1][c * 4]);
            }
        }

        // ---- h @ W_h^T, double-buffered A fragments ----
        const bf16_t* arow = Hcur + (size_t)(mbase + l15) * kEMB;
        v8f acc = {};
        v16bf a0 = load_frag_a(arow, 0, hi);
#pragma unroll 4
        for (int k0 = 32; k0 < kEMB; k0 += 32) {
            const v16bf a1 = load_frag_a(arow, k0, hi);
            const v16bf bb = load_frag_b(bptr, k0 - 32, hi);
            acc = wmma_bf16(a0, bb, acc);
            a0 = a1;
        }
        acc = wmma_bf16(a0, load_frag_b(bptr, kEMB - 32, hi), acc);

        // ---- epilogue: + xw (from LDS), tanh, store bf16 h and hs ----
        const float* xwt = lds_xw[cur];
#pragma unroll
        for (int v = 0; v < 8; ++v) {
            const int br = mbase + hi * 8 + v;                 // batch row
            const float val = acc[v] + xwt[br * 32 + nloc];
            const bf16_t hb = (bf16_t)tanhf(val);
            Hnxt[(size_t)br * kEMB + n] = hb;
            hs[((size_t)br * kT + t) * kEMB + n] = hb;
        }

        // ---- grid-wide step barrier (release writes, acquire others') ----
        __threadfence();
        __syncthreads();
        if (threadIdx.x == 0) {
            __hip_atomic_fetch_add(barrier_cnt, 1u, __ATOMIC_RELEASE,
                                   __HIP_MEMORY_SCOPE_AGENT);
            const unsigned target = (unsigned)(t + 1) * kNWG;
            while (__hip_atomic_load(barrier_cnt, __ATOMIC_ACQUIRE,
                                     __HIP_MEMORY_SCOPE_AGENT) < target) {
                __builtin_amdgcn_s_sleep(1);
            }
        }
        wait_async_lds();       // next xw tile resident before next iteration
        __syncthreads();
        __threadfence();
        bf16_t* tmp = Hcur; Hcur = Hnxt; Hnxt = tmp;
        cur ^= 1;
    }
}

// ---------------------------------------------------------------------------
// Kernel 3: out[b*1024+t][o] = sum_e hs[..][e] * w_head[o][e] + b_head[o]
// ---------------------------------------------------------------------------
__global__ __launch_bounds__(256) void head_gemm_kernel(
    const bf16_t* __restrict__ hs, const bf16_t* __restrict__ wheadb,
    const float* __restrict__ bhead, float* __restrict__ out) {
    const int wave = threadIdx.x >> 5;
    const int lane = threadIdx.x & 31;
    const int l15 = lane & 15, hi = lane >> 4;
    const int mtile = blockIdx.y * 128 + wave * 16;
    const int ntile = blockIdx.x * 64;
    const bf16_t* arow  = hs + (size_t)(mtile + l15) * kEMB;
    const bf16_t* brow0 = wheadb + (size_t)(ntile + l15) * kEMB;

    v8f acc[4] = {};
    v16bf a0 = load_frag_a(arow, 0, hi);
    v16bf b0[4];
#pragma unroll
    for (int j = 0; j < 4; ++j) b0[j] = load_frag_b(brow0 + (size_t)j * 16 * kEMB, 0, hi);

#pragma unroll
    for (int k0 = 32; k0 < kEMB; k0 += 32) {
        const v16bf a1 = load_frag_a(arow, k0, hi);
        v16bf b1[4];
#pragma unroll
        for (int j = 0; j < 4; ++j)
            b1[j] = load_frag_b(brow0 + (size_t)j * 16 * kEMB, k0, hi);
#pragma unroll
        for (int j = 0; j < 4; ++j) acc[j] = wmma_bf16(a0, b0[j], acc[j]);
        a0 = a1;
#pragma unroll
        for (int j = 0; j < 4; ++j) b0[j] = b1[j];
    }
#pragma unroll
    for (int j = 0; j < 4; ++j) acc[j] = wmma_bf16(a0, b0[j], acc[j]);

#pragma unroll
    for (int j = 0; j < 4; ++j) {
        const int n = ntile + j * 16 + l15;
        const float bias = bhead[n];
#pragma unroll
        for (int v = 0; v < 8; ++v) {
            const int mm = mtile + hi * 8 + v;
            out[(size_t)mm * kVOCAB + n] = acc[j][v] + bias;
        }
    }
}

// ---------------------------------------------------------------------------
extern "C" void kernel_launch(void* const* d_in, const int* in_sizes, int n_in,
                              void* d_out, int out_size, void* d_ws, size_t ws_size,
                              hipStream_t stream) {
    const int*   x      = (const int*)  d_in[0];
    const float* emb    = (const float*)d_in[1];
    const float* w_cell = (const float*)d_in[2];
    const float* b_cell = (const float*)d_in[3];
    const float* w_head = (const float*)d_in[4];
    const float* b_head = (const float*)d_in[5];
    const float* h0     = (const float*)d_in[6];
    float* out = (float*)d_out;

    // Workspace layout (~203 MB total)
    char* ws = (char*)d_ws;
    size_t off = 0;
    float*  xw     = (float*) (ws + off); off += (size_t)kT * kB * kEMB * 4;      // 128 MB
    bf16_t* hsbuf  = (bf16_t*)(ws + off); off += (size_t)kB * kT * kEMB * 2;      //  64 MB
    bf16_t* embb   = (bf16_t*)(ws + off); off += (size_t)kVOCAB * kEMB * 2;
    bf16_t* wx     = (bf16_t*)(ws + off); off += (size_t)kEMB * kEMB * 2;
    bf16_t* wh     = (bf16_t*)(ws + off); off += (size_t)kEMB * kEMB * 2;
    bf16_t* wheadb = (bf16_t*)(ws + off); off += (size_t)kVOCAB * kEMB * 2;
    bf16_t* Hbuf   = (bf16_t*)(ws + off); off += (size_t)2 * kB * kEMB * 2;       // ping-pong
    unsigned* barrier_cnt = (unsigned*)(ws + off); off += 256;

    prep_kernel<<<256, 256, 0, stream>>>(emb, w_cell, w_head, h0,
                                         embb, wx, wh, wheadb, Hbuf, barrier_cnt);

    dim3 g1(kEMB / 64, kRows / 128);   // (8, 512)
    xw_gemm_kernel<<<g1, 256, 0, stream>>>(x, embb, wx, b_cell, xw);

    rnn_scan_kernel<<<kNWG, 256, 0, stream>>>(wh, xw, Hbuf, hsbuf, barrier_cnt);

    dim3 g2(kVOCAB / 64, kRows / 128); // (4, 512)
    head_gemm_kernel<<<g2, 256, 0, stream>>>(hsbuf, wheadb, b_head, out);
}
